// LocalPatternExpert_56633438765491
// MI455X (gfx1250) — matrix operand order repaired and checked by reading, not compiled, not run on hardware
//
#include <hip/hip_runtime.h>
#include <hip/hip_bf16.h>

// ---------------------------------------------------------------------------
// CDNA5 (gfx1250) windowed-attention transformer block.
// All matmuls use v_wmma_f32_16x16x32_bf16 (wave32, f32 accumulate).
// Operands are pre-packed into WMMA fragment layouts (contiguous k order, same in A and B):
//   A 16x32 bf16 : lane = half*16+m, elem e -> K=half*16+e
//   B 32x16 bf16 : lane = half*16+n, elem e -> K=half*16+e
//   C 16x16 f32  : lane = half*16+n, elem r -> M=half*8+r
// so each fragment is ONE contiguous 32-byte load per lane.
// GEMMs use 4x2 register blocking: 8 wmma per 6 fragment loads per K-step.
// ---------------------------------------------------------------------------

typedef __attribute__((ext_vector_type(16))) __bf16 v16bf;
typedef __attribute__((ext_vector_type(8)))  float  v8f;

#define S_LEN   2048
#define DMODEL  1024
#define NHEAD   16
#define HDIM    64
#define WIN     256

static __device__ __forceinline__ unsigned short f2bf_u16(float f) {
  unsigned int u = __builtin_bit_cast(unsigned int, f);
  unsigned int r = u + 0x7FFFu + ((u >> 16) & 1u);   // round-to-nearest-even
  return (unsigned short)(r >> 16);
}
static __device__ __forceinline__ __bf16 f2bf(float f) {
  unsigned short s = f2bf_u16(f);
  return __builtin_bit_cast(__bf16, s);
}
static __device__ __forceinline__ __bf16 u2bf(unsigned short s) {
  return __builtin_bit_cast(__bf16, s);
}
// K offset within a 16x32 A fragment for (elem e, lane half)
static __device__ __forceinline__ int a_koff(int e, int half) {
  return e + half * 16;
}

// --------------------------- operand packing -------------------------------
// packed A: [(mtile*KT + kt)*32 + lane]*16 + e   (M/16 x K/32 fragments)
__global__ void pack_a_kernel(const float* __restrict__ src, __bf16* __restrict__ dst,
                              int M, int K) {
  int idx = blockIdx.x * blockDim.x + threadIdx.x;
  if (idx >= M * K) return;
  int e    = idx & 15;
  int lane = (idx >> 4) & 31;
  int tile = idx >> 9;
  int KT   = K >> 5;
  int kt    = tile % KT;
  int mtile = tile / KT;
  int half = lane >> 4;
  int m    = lane & 15;
  int k    = kt * 32 + a_koff(e, half);
  dst[idx] = f2bf(src[(size_t)(mtile * 16 + m) * K + k]);
}

// packed B: [(kt*NT + ntile)*32 + lane]*16 + e   (K/32 x N/16 fragments)
__global__ void pack_b_kernel(const float* __restrict__ src, __bf16* __restrict__ dst,
                              int K, int N) {
  int idx = blockIdx.x * blockDim.x + threadIdx.x;
  if (idx >= K * N) return;
  int e    = idx & 15;
  int lane = (idx >> 4) & 31;
  int tile = idx >> 9;
  int NT   = N >> 4;
  int ntile = tile % NT;
  int kt    = tile / NT;
  int n = ntile * 16 + (lane & 15);
  int k = kt * 32 + (lane >> 4) * 16 + e;
  dst[idx] = f2bf(src[(size_t)k * N + n]);
}

// --------------------------- WMMA GEMM (4x2 blocked) -----------------------
// one wave -> one 64x32 macro-tile (4x2 grid of 16x16 WMMA tiles).
// Per K-step: 4 A-fragments + 2 B-fragments loaded, 8 v_wmma issued.
template <bool ADD_RES>
__global__ void gemm_bf16_kernel(const __bf16* __restrict__ pA,
                                 const __bf16* __restrict__ pB,
                                 const float* __restrict__ resid,
                                 float* __restrict__ C,
                                 int MB, int NB, int KT, int N) {
  int wave = blockIdx.x * (blockDim.x >> 5) + (threadIdx.x >> 5);
  int lane = threadIdx.x & 31;
  int mb = wave / NB;   // 64-row block
  int nb = wave % NB;   // 32-col block
  if (mb >= MB) return;

  const v16bf* pa = reinterpret_cast<const v16bf*>(pA);
  const v16bf* pb = reinterpret_cast<const v16bf*>(pB);
  int NT = NB * 2;

  v8f acc[4][2];
#pragma unroll
  for (int i = 0; i < 4; ++i)
#pragma unroll
    for (int j = 0; j < 2; ++j)
#pragma unroll
      for (int r = 0; r < 8; ++r) acc[i][j][r] = 0.f;

  for (int kt = 0; kt < KT; ++kt) {
    // prefetch a few K-steps ahead (lowers to global_prefetch_b8)
    int ktp = (kt + 6 < KT) ? kt + 6 : KT - 1;
    __builtin_prefetch((const void*)&pa[(size_t)((mb * 4) * KT + ktp) * 32 + lane], 0, 1);
    __builtin_prefetch((const void*)&pb[(size_t)(ktp * NT + nb * 2) * 32 + lane], 0, 1);

    v16bf a[4], b[2];
#pragma unroll
    for (int i = 0; i < 4; ++i)
      a[i] = pa[(size_t)((mb * 4 + i) * KT + kt) * 32 + lane];
#pragma unroll
    for (int j = 0; j < 2; ++j)
      b[j] = pb[(size_t)(kt * NT + nb * 2 + j) * 32 + lane];
#pragma unroll
    for (int i = 0; i < 4; ++i)
#pragma unroll
      for (int j = 0; j < 2; ++j)
        acc[i][j] = __builtin_amdgcn_wmma_f32_16x16x32_bf16(
            false, a[i], false, b[j], (short)0, acc[i][j], false, false);
  }

  int half = lane >> 4;
#pragma unroll
  for (int i = 0; i < 4; ++i)
#pragma unroll
    for (int j = 0; j < 2; ++j) {
      int n = (nb * 2 + j) * 16 + (lane & 15);
#pragma unroll
      for (int r = 0; r < 8; ++r) {
        int row = (mb * 4 + i) * 16 + r + half * 8;
        float v = acc[i][j][r];
        if (ADD_RES) v += resid[(size_t)row * N + n];
        C[(size_t)row * N + n] = v;
      }
    }
}

// --------------------------- RoPE + head split -----------------------------
// qkv f32 [S,3072] -> q,k bf16 [H,S,64] (rope applied), V^T bf16 [H,64,S]
__global__ void rope_split_kernel(const float* __restrict__ qkv,
                                  __bf16* __restrict__ qbf,
                                  __bf16* __restrict__ kbf,
                                  __bf16* __restrict__ vt) {
  int h = blockIdx.y;
  int s = blockIdx.x * 4 + (threadIdx.x >> 6);
  int d = threadIdx.x & 63;
  const float* row = qkv + (size_t)s * 3072;
  int col = h * 64 + d;
  float q = row[col];
  float k = row[1024 + col];
  float v = row[2048 + col];
  int i = d & 31;
  float inv = __expf(-(float)i * (9.210340371976184f / 32.0f));  // 10000^(-i/32)
  float ang = (float)s * inv;
  float sn, c;
  __sincosf(ang, &sn, &c);
  int pcol = h * 64 + ((d < 32) ? d + 32 : d - 32);
  float qp = row[pcol], kp = row[1024 + pcol];
  float qr = (d < 32) ? -qp : qp;
  float kr = (d < 32) ? -kp : kp;
  size_t hs = ((size_t)h * S_LEN + s) * 64 + d;
  qbf[hs] = f2bf(q * c + qr * sn);
  kbf[hs] = f2bf(k * c + kr * sn);
  vt[((size_t)h * 64 + d) * S_LEN + s] = f2bf(v);
}

// --------------------------- windowed flash attention ----------------------
// one 32-thread wave per (head, 16-query tile). Online softmax over 32-key
// blocks; P is round-tripped through LDS to convert C-layout -> A-layout.
// Output written directly in packed-A layout for the out-proj GEMM (K=1024).
__global__ __launch_bounds__(32)
void attn_kernel(const __bf16* __restrict__ qbf, const __bf16* __restrict__ kbf,
                 const __bf16* __restrict__ vt, __bf16* __restrict__ packedA) {
  __shared__ unsigned short ldsP[16 * 32];
  int h    = blockIdx.y;
  int s0   = blockIdx.x * 16;
  int lane = threadIdx.x;
  int half = lane >> 4;
  int lm   = lane & 15;

  // Q as two A fragments (dims 0..31, 32..63)
  v16bf qa[2];
  {
    const __bf16* qrow = qbf + ((size_t)h * S_LEN + (s0 + lm)) * 64;
#pragma unroll
    for (int kt = 0; kt < 2; ++kt)
#pragma unroll
      for (int e = 0; e < 16; ++e)
        qa[kt][e] = qrow[kt * 32 + a_koff(e, half)];
  }

  v8f acc[4];
  float mrun[8], lrun[8];
#pragma unroll
  for (int nt = 0; nt < 4; ++nt)
#pragma unroll
    for (int r = 0; r < 8; ++r) acc[nt][r] = 0.f;
#pragma unroll
  for (int r = 0; r < 8; ++r) { mrun[r] = -1e30f; lrun[r] = 0.f; }

  int lo  = s0 - WIN; if (lo < 0) lo = 0;
  int kb0 = lo & ~31;
  for (int kb = kb0; kb <= s0 + 15; kb += 32) {
    // ---- scores = Q @ K^T : two 16x16 tiles (keys kb..kb+15, kb+16..kb+31)
    v8f sc[2];
#pragma unroll
    for (int nt = 0; nt < 2; ++nt) {
#pragma unroll
      for (int r = 0; r < 8; ++r) sc[nt][r] = 0.f;
      int t = kb + nt * 16 + lm;
#pragma unroll
      for (int kt = 0; kt < 2; ++kt) {
        v16bf bv = *reinterpret_cast<const v16bf*>(
            kbf + ((size_t)h * S_LEN + t) * 64 + kt * 32 + half * 16);
        sc[nt] = __builtin_amdgcn_wmma_f32_16x16x32_bf16(false, qa[kt], false, bv,
                                                         (short)0, sc[nt], false, false);
      }
    }
    // ---- online softmax, row-wise (row r+half*8 spans 16 lanes of a half)
#pragma unroll
    for (int r = 0; r < 8; ++r) {
      int srow = s0 + r + half * 8;
      int t0 = kb + lm, t1 = kb + 16 + lm;
      float v0 = (t0 <= srow && srow - t0 <= WIN) ? sc[0][r] * 0.125f : -1e30f;
      float v1 = (t1 <= srow && srow - t1 <= WIN) ? sc[1][r] * 0.125f : -1e30f;
      float mx = fmaxf(v0, v1);
      mx = fmaxf(mx, __shfl_xor(mx, 1, 16));
      mx = fmaxf(mx, __shfl_xor(mx, 2, 16));
      mx = fmaxf(mx, __shfl_xor(mx, 4, 16));
      mx = fmaxf(mx, __shfl_xor(mx, 8, 16));
      float mnew  = fmaxf(mrun[r], mx);
      float scale = __expf(mrun[r] - mnew);
      float p0 = __expf(v0 - mnew);
      float p1 = __expf(v1 - mnew);
      float rs = p0 + p1;
      rs += __shfl_xor(rs, 1, 16);
      rs += __shfl_xor(rs, 2, 16);
      rs += __shfl_xor(rs, 4, 16);
      rs += __shfl_xor(rs, 8, 16);
      lrun[r] = lrun[r] * scale + rs;
      mrun[r] = mnew;
#pragma unroll
      for (int nt = 0; nt < 4; ++nt) acc[nt][r] *= scale;
      int m = r + half * 8;
      ldsP[m * 32 + lm]      = f2bf_u16(p0);
      ldsP[m * 32 + 16 + lm] = f2bf_u16(p1);
    }
    __syncthreads();
    // ---- P (16x32) back as an A fragment
    v16bf pa;
#pragma unroll
    for (int e = 0; e < 16; ++e)
      pa[e] = u2bf(ldsP[lm * 32 + a_koff(e, half)]);
    __syncthreads();
    // ---- acc += P @ V  (V^T layout gives contiguous B fragments)
#pragma unroll
    for (int nt = 0; nt < 4; ++nt) {
      int dim = nt * 16 + lm;
      v16bf bv = *reinterpret_cast<const v16bf*>(
          vt + ((size_t)h * 64 + dim) * S_LEN + kb + half * 16);
      acc[nt] = __builtin_amdgcn_wmma_f32_16x16x32_bf16(false, pa, false, bv,
                                                        (short)0, acc[nt], false, false);
    }
  }

  // ---- normalize and scatter straight into packed-A layout (K=1024, KT=32)
  int mtile = blockIdx.x;
#pragma unroll
  for (int r = 0; r < 8; ++r) {
    float invl = 1.0f / lrun[r];
    int m = r + half * 8;
#pragma unroll
    for (int nt = 0; nt < 4; ++nt) {
      float v = acc[nt][r] * invl;
      int kfeat = h * 64 + nt * 16 + lm;
      int ktile = kfeat >> 5;
      int koff  = kfeat & 31;
      int half2 = koff >> 4;
      int e2    = koff & 15;
      int lane2 = m + 16 * half2;
      packedA[(((size_t)mtile * 32 + ktile) * 32 + lane2) * 16 + e2] = f2bf(v);
    }
  }
}

// --------------------------- residual LayerNorm ----------------------------
__global__ void ln_kernel(const float* __restrict__ res, const float* __restrict__ gamma,
                          const float* __restrict__ beta, float* __restrict__ out) {
  __shared__ float sbuf[16];
  int row = blockIdx.x;
  int tid = threadIdx.x;  // 256
  const float* r = res + (size_t)row * DMODEL;
  float s = 0.f, s2 = 0.f, vals[4];
#pragma unroll
  for (int i = 0; i < 4; ++i) {
    float v = r[tid + i * 256];
    vals[i] = v; s += v; s2 += v * v;
  }
#pragma unroll
  for (int off = 16; off > 0; off >>= 1) {
    s  += __shfl_down(s, off, 32);
    s2 += __shfl_down(s2, off, 32);
  }
  int wid = tid >> 5;
  if ((tid & 31) == 0) { sbuf[wid] = s; sbuf[8 + wid] = s2; }
  __syncthreads();
  if (tid == 0) {
    float ts = 0.f, ts2 = 0.f;
    for (int i = 0; i < 8; ++i) { ts += sbuf[i]; ts2 += sbuf[8 + i]; }
    sbuf[0] = ts; sbuf[8] = ts2;
  }
  __syncthreads();
  float mu   = sbuf[0] * (1.0f / DMODEL);
  float var  = sbuf[8] * (1.0f / DMODEL) - mu * mu;
  float rstd = rsqrtf(var + 1e-5f);
#pragma unroll
  for (int i = 0; i < 4; ++i) {
    int c = tid + i * 256;
    out[(size_t)row * DMODEL + c] = (vals[i] - mu) * rstd * gamma[c] + beta[c];
  }
}

__global__ void copy_state_kernel(const float* __restrict__ st, float* __restrict__ out) {
  int i = blockIdx.x * blockDim.x + threadIdx.x;
  if (i < DMODEL) out[(size_t)S_LEN * DMODEL + i] = st[i];
}

// --------------------------- launcher --------------------------------------
extern "C" void kernel_launch(void* const* d_in, const int* in_sizes, int n_in,
                              void* d_out, int out_size, void* d_ws, size_t ws_size,
                              hipStream_t stream) {
  const float* x     = (const float*)d_in[0];
  const float* state = (const float*)d_in[1];
  const float* Wqkv  = (const float*)d_in[2];
  const float* Wout  = (const float*)d_in[3];
  const float* gamma = (const float*)d_in[4];
  const float* beta  = (const float*)d_in[5];
  float* out = (float*)d_out;

  char* ws = (char*)d_ws;
  size_t off = 0;
  __bf16* pAx    = (__bf16*)(ws + off); off += (size_t)S_LEN * DMODEL * 2;       // 4 MB
  __bf16* pBqkv  = (__bf16*)(ws + off); off += (size_t)DMODEL * 3 * DMODEL * 2;  // 6 MB
  __bf16* pBout  = (__bf16*)(ws + off); off += (size_t)DMODEL * DMODEL * 2;      // 2 MB
  float*  qkv    = (float*)(ws + off);  off += (size_t)S_LEN * 3 * DMODEL * 4;   // 24 MB
  __bf16* qbf    = (__bf16*)(ws + off); off += (size_t)NHEAD * S_LEN * HDIM * 2; // 4 MB
  __bf16* kbf    = (__bf16*)(ws + off); off += (size_t)NHEAD * S_LEN * HDIM * 2; // 4 MB
  __bf16* vt     = (__bf16*)(ws + off); off += (size_t)NHEAD * HDIM * S_LEN * 2; // 4 MB
  __bf16* pAattn = (__bf16*)(ws + off); off += (size_t)S_LEN * DMODEL * 2;       // 4 MB
  float*  res    = (float*)(ws + off);  off += (size_t)S_LEN * DMODEL * 4;       // 8 MB

  // pack operands into WMMA fragment layouts
  pack_a_kernel<<<(S_LEN * DMODEL) / 256, 256, 0, stream>>>(x, pAx, S_LEN, DMODEL);
  pack_b_kernel<<<(DMODEL * 3 * DMODEL) / 256, 256, 0, stream>>>(Wqkv, pBqkv, DMODEL, 3 * DMODEL);
  pack_b_kernel<<<(DMODEL * DMODEL) / 256, 256, 0, stream>>>(Wout, pBout, DMODEL, DMODEL);

  // qkv = x @ W_qkv           (MB=32 (64-row), NB=96 (32-col), KT=32)
  gemm_bf16_kernel<false><<<(32 * 96) / 8, 256, 0, stream>>>(
      pAx, pBqkv, nullptr, qkv, 32, 96, 32, 3 * DMODEL);

  // RoPE + head split + V transpose
  rope_split_kernel<<<dim3(S_LEN / 4, NHEAD), 256, 0, stream>>>(qkv, qbf, kbf, vt);

  // windowed flash attention -> packed-A for out-proj
  attn_kernel<<<dim3(S_LEN / 16, NHEAD), 32, 0, stream>>>(qbf, kbf, vt, pAattn);

  // res = attn_out @ W_out + x   (MB=32, NB=32, KT=32)
  gemm_bf16_kernel<true><<<(32 * 32) / 8, 256, 0, stream>>>(
      pAattn, pBout, x, res, 32, 32, 32, DMODEL);

  // y = LayerNorm(res) ; state passthrough
  ln_kernel<<<S_LEN, 256, 0, stream>>>(res, gamma, beta, out);
  copy_state_kernel<<<4, 256, 0, stream>>>(state, out);
}
